// NeuralPredictorModel_3994319585307
// MI455X (gfx1250) — matrix operand order repaired and checked by reading, not compiled, not run on hardware
//
#include <hip/hip_runtime.h>
#include <hip/hip_bf16.h>

// GCN neural predictor on gfx1250 (MI455X), wave32 + v_wmma_f32_16x16x32_f16.
// B=65536 graphs, GS=7 nodes, IN_H=5, GCN_H=144, LIN_H=128.
// ~85 GFLOP vs ~22 MB traffic -> compute bound -> all GEMMs via WMMA (f16 in, f32 acc).
// Weights pre-packed to f16 (transposed, zero-padded) in d_ws once per call, then
// staged to LDS per layer via GLOBAL_LOAD_ASYNC_TO_LDS_B128 (ASYNCcnt path).
// Intra-wave LDS round-trips use wavefront fences only (LDS is in-order per wave).

typedef _Float16 h8  __attribute__((ext_vector_type(8)));
typedef _Float16 h16 __attribute__((ext_vector_type(16)));
typedef float    f8  __attribute__((ext_vector_type(8)));
typedef int      v4i __attribute__((ext_vector_type(4)));
typedef __attribute__((address_space(1))) v4i* gv4p;   // global int4*
typedef __attribute__((address_space(3))) v4i* lv4p;   // LDS int4*

#define WAVES   4
#define THREADS (WAVES * 32)
#define BPW     2                 // batches per wave (2*7=14 rows -> 16-row M tile)
#define BPG     (WAVES * BPW)     // 8 batches per workgroup

#define GCN_H   144
#define KPAD    160               // 144 padded to 5 chunks of 32
#define SIDE_ELEMS (GCN_H * KPAD) // 23040 halfs per weight side
#define LAYER_ELEMS (2 * SIDE_ELEMS)
#define PACK_ELEMS  (3 * LAYER_ELEMS)   // 138240 halfs in d_ws

// ---- dynamic LDS layout (bytes) ----
#define WT_OFF   0
#define WT_BYTES (LAYER_ELEMS * 2)                     // 92160: w1^T and w2^T, f16
#define XB_OFF   (WT_OFF + WT_BYTES)                   // per-wave x: [16][160] f16
#define XB_ELEMS (16 * KPAD)
#define YB_OFF   (XB_OFF + WAVES * XB_ELEMS * 2)       // per-wave yT: 2 slots [16][32] f16
#define YB_ELEMS (2 * 16 * 32)
#define ADJ_OFF  (YB_OFF + WAVES * YB_ELEMS * 2)       // per-wave: adj_d[2][7][7] + colsum[2][7]
#define ADJ_FLOATS 112
#define WEFF_OFF (ADJ_OFF + WAVES * ADJ_FLOATS * 4)    // folded head weight: 144 f32
#define SMEM_BYTES (WEFF_OFF + 144 * 4)                // 123200 B < 320 KB WGP LDS

__device__ __forceinline__ h16 cat8(h8 a, h8 b) {
  return __builtin_shufflevector(a, b, 0,1,2,3,4,5,6,7,8,9,10,11,12,13,14,15);
}

// Intra-wave LDS ordering: hardware keeps per-wave LDS ops in order (ISA 7.3),
// so cross-lane store->load inside one wave only needs a compiler fence.
__device__ __forceinline__ void wfence() {
  __builtin_amdgcn_fence(__ATOMIC_ACQ_REL, "wavefront");
  __builtin_amdgcn_wave_barrier();
}

__device__ __forceinline__ void wait_async0() {
#if __has_builtin(__builtin_amdgcn_s_wait_asynccnt)
  __builtin_amdgcn_s_wait_asynccnt(0);
#else
  asm volatile("s_wait_asynccnt 0" ::: "memory");
#endif
}

// ---- prologue: pack all GCN weights as f16, transposed [l][side][n][k<KPAD] ----
__global__ __launch_bounds__(256)
void pack_weights_kernel(const float* __restrict__ w1_0, const float* __restrict__ w2_0,
                         const float* __restrict__ w1_1, const float* __restrict__ w2_1,
                         const float* __restrict__ w1_2, const float* __restrict__ w2_2,
                         _Float16* __restrict__ wpack)
{
  int idx = blockIdx.x * 256 + threadIdx.x;
  if (idx >= PACK_ELEMS) return;
  int l    = idx / LAYER_ELEMS;
  int r    = idx % LAYER_ELEMS;
  int side = r / SIDE_ELEMS;
  int e    = r % SIDE_ELEMS;
  int n    = e / KPAD;
  int k    = e % KPAD;
  const float* w;
  int K;
  if (l == 0)      { w = side ? w2_0 : w1_0; K = 5;   }
  else if (l == 1) { w = side ? w2_1 : w1_1; K = 144; }
  else             { w = side ? w2_2 : w1_2; K = 144; }
  wpack[idx] = (_Float16)((k < K) ? w[k * GCN_H + n] : 0.f);
}

__global__ __launch_bounds__(THREADS)
void gcn_wmma_kernel(const float* __restrict__ ops,
                     const float* __restrict__ adj,
                     const float* __restrict__ numv,
                     const _Float16* __restrict__ wpack,
                     const float* __restrict__ fc1,  const float* __restrict__ fc2,
                     float* __restrict__ out)
{
  extern __shared__ __attribute__((aligned(16))) char smem[];
  _Float16* wT   = (_Float16*)(smem + WT_OFF);
  const int tid  = threadIdx.x;
  const int wave = tid >> 5;
  const int lane = tid & 31;
  _Float16* xbuf = (_Float16*)(smem + XB_OFF) + wave * XB_ELEMS;   // [m][k]
  _Float16* ybuf = (_Float16*)(smem + YB_OFF) + wave * YB_ELEMS;   // 2 slots [col][node]
  float*    adjd = (float*)(smem + ADJ_OFF) + wave * ADJ_FLOATS;   // [2][7][7] + colsum[2][7]
  float*    weff = (float*)(smem + WEFF_OFF);
  const int wbase = blockIdx.x * BPG + wave * BPW;                 // first batch of this wave
  const int m16 = lane & 15;
  const int h   = lane >> 4;

  // ---- zero per-wave LDS buffers (padding rows/cols must read as 0) ----
  h8 z8 = {};
  for (int i = lane; i < XB_ELEMS / 8; i += 32) ((h8*)xbuf)[i] = z8;
  for (int i = lane; i < YB_ELEMS / 8; i += 32) ((h8*)ybuf)[i] = z8;

  // ---- folded head: out = pooled . (fc1^T @ fc2)  (exact linear collapse) ----
  for (int k = tid; k < GCN_H; k += THREADS) {
    float s = 0.f;
    for (int o = 0; o < 128; ++o) s += fc1[o * GCN_H + k] * fc2[o];
    weff[k] = s;
  }

  // ---- adjacency prep: adj_d = row_norm(adj + I); colsums for transposed norm ----
  if (lane < 14) {
    int b = lane / 7, i = lane % 7;
    const float* ar = adj + (size_t)(wbase + b) * 49 + i * 7;
    float a[7]; float rs = 0.f;
#pragma unroll
    for (int j = 0; j < 7; ++j) { a[j] = ar[j] + (i == j ? 1.f : 0.f); rs += a[j]; }
    float inv = 1.f / rs;
#pragma unroll
    for (int j = 0; j < 7; ++j) adjd[b * 49 + i * 7 + j] = a[j] * inv;
  }
  wfence();
  if (lane < 14) {
    int b = lane / 7, i = lane % 7;
    float cs = 0.f;
#pragma unroll
    for (int j = 0; j < 7; ++j) cs += adjd[b * 49 + j * 7 + i];
    adjd[98 + b * 7 + i] = cs;
  }
  // ---- layer-0 features into xbuf (k<5 valid, rest stays zero) ----
  if (lane < 14) {
    int b = lane / 7, i = lane % 7;
    const float* op = ops + (size_t)(wbase + b) * 35 + i * 5;
#pragma unroll
    for (int k = 0; k < 5; ++k) xbuf[(b * 7 + i) * KPAD + k] = (_Float16)op[k];
  }
  wfence();

  // ---- constant adjacency A-fragments (block-diagonal 2x [7x7] in 16x32, f16) ----
  // A-layout: element e of lane -> K = (e/8)*16 + (lane/16)*8 + e%8, M = lane%16
  h16 A1f, A2f;
#pragma unroll
  for (int e = 0; e < 16; ++e) {
    int j = (e >> 3) * 16 + h * 8 + (e & 7);
    float v1 = 0.f, v2 = 0.f;
    if (m16 < 14 && j < 14 && (m16 / 7) == (j / 7)) {
      int b = m16 / 7, i = m16 % 7, jj = j % 7;
      v1 = adjd[b * 49 + i * 7 + jj];                        // row-normalized adj_d
      v2 = adjd[b * 49 + jj * 7 + i] / adjd[98 + b * 7 + i]; // row_norm(adj_d^T)
    }
    A1f[e] = (_Float16)v1;
    A2f[e] = (_Float16)v2;
  }

  // ---- 3 GCN layers ----
  for (int l = 0; l < 3; ++l) {
    __syncthreads();                       // all waves done with previous wT
    // stage this layer's packed f16 weights into LDS (async DMA path)
    const _Float16* wsrc = wpack + (size_t)l * LAYER_ELEMS;
#if __has_builtin(__builtin_amdgcn_global_load_async_to_lds_b128)
    for (int i = tid; i < LAYER_ELEMS / 8; i += THREADS) {
      __builtin_amdgcn_global_load_async_to_lds_b128(
          (gv4p)(wsrc + i * 8), (lv4p)(wT + i * 8), 0, 0);
    }
    wait_async0();
#else
    for (int i = tid; i < LAYER_ELEMS / 8; i += THREADS)
      ((h8*)wT)[i] = ((const h8*)wsrc)[i];
#endif
    __syncthreads();

    // preload X A-fragments (per-lane: two contiguous b128 LDS loads each)
    h16 xa[5];
#pragma unroll
    for (int c = 0; c < 5; ++c) {
      const _Float16* p = xbuf + m16 * KPAD + c * 32 + h * 8;
      xa[c] = cat8(*(const h8*)p, *(const h8*)(p + 16));
    }

    for (int t = 0; t < 9; ++t) {          // 9 N-tiles of 16 over GCN_H=144
      const int ncol = t * 16 + m16;
      f8 acc1 = {}, acc2 = {};
#pragma unroll
      for (int c = 0; c < 5; ++c) {        // K = 5 chunks of 32
        // B-layout: element e -> K = (e/8)*16 + (lane/16)*8 + e%8, N = lane%16
        const _Float16* wp1 = wT + ncol * KPAD + c * 32 + h * 8;
        h16 b1 = cat8(*(const h8*)wp1, *(const h8*)(wp1 + 16));
        acc1 = __builtin_amdgcn_wmma_f32_16x16x32_f16(false, xa[c], false, b1,
                                                      (short)0, acc1, false, false);
        const _Float16* wp2 = wp1 + SIDE_ELEMS;
        h16 b2 = cat8(*(const h8*)wp2, *(const h8*)(wp2 + 16));
        acc2 = __builtin_amdgcn_wmma_f32_16x16x32_f16(false, xa[c], false, b2,
                                                      (short)0, acc2, false, false);
      }

      // stage both Y tiles (D layout -> transposed per-wave LDS, one slot per tile)
      { h8 s1, s2;
#pragma unroll
        for (int r = 0; r < 8; ++r) {      // D: M = r + 8h, N = m16
          s1[r] = (_Float16)acc1[r];
          s2[r] = (_Float16)acc2[r];
        }
        *(h8*)(ybuf + m16 * 32 + h * 8)       = s1;
        *(h8*)(ybuf + 512 + m16 * 32 + h * 8) = s2;
      }
      wfence();                            // intra-wave LDS is in-order; fence compiler only
      h16 yb1 = cat8(*(const h8*)(ybuf + m16 * 32 + h * 8),
                     *(const h8*)(ybuf + m16 * 32 + h * 8 + 16));
      h16 yb2 = cat8(*(const h8*)(ybuf + 512 + m16 * 32 + h * 8),
                     *(const h8*)(ybuf + 512 + m16 * 32 + h * 8 + 16));
      f8 zc = {};
      f8 o1 = __builtin_amdgcn_wmma_f32_16x16x32_f16(false, A1f, false, yb1,
                                                     (short)0, zc, false, false);
      f8 o2 = __builtin_amdgcn_wmma_f32_16x16x32_f16(false, A2f, false, yb2,
                                                     (short)0, zc, false, false);

      // x_new = 0.5*(relu(o1)+relu(o2)); write back to xbuf (A-frags already in regs)
#pragma unroll
      for (int r = 0; r < 8; ++r) {
        float v = 0.5f * (fmaxf((float)o1[r], 0.f) + fmaxf((float)o2[r], 0.f));
        xbuf[(r + 8 * h) * KPAD + ncol] = (_Float16)v;
      }
      wfence();                            // keep next iter's ybuf stores behind these loads
    }
  }
  __syncthreads();

  // ---- head: out[b] = (sum_nodes x[b]) . weff / num_vertices[b] ----
  float a0 = 0.f, a1 = 0.f;
  for (int k = lane; k < GCN_H; k += 32) {
    float we = weff[k];
    float s0 = 0.f, s1 = 0.f;
#pragma unroll
    for (int n = 0; n < 7; ++n) {
      s0 += (float)xbuf[n * KPAD + k];
      s1 += (float)xbuf[(7 + n) * KPAD + k];
    }
    a0 += s0 * we;
    a1 += s1 * we;
  }
#pragma unroll
  for (int off = 16; off > 0; off >>= 1) {
    a0 += __shfl_xor(a0, off, 32);
    a1 += __shfl_xor(a1, off, 32);
  }
  if (lane == 0) {
    out[wbase]     = a0 / numv[wbase];
    out[wbase + 1] = a1 / numv[wbase + 1];
  }
}

extern "C" void kernel_launch(void* const* d_in, const int* in_sizes, int n_in,
                              void* d_out, int out_size, void* d_ws, size_t ws_size,
                              hipStream_t stream) {
  const float* ops  = (const float*)d_in[0];
  const float* adjm = (const float*)d_in[1];
  const float* numv = (const float*)d_in[2];
  const float* w1_0 = (const float*)d_in[3];
  const float* w2_0 = (const float*)d_in[4];
  const float* w1_1 = (const float*)d_in[5];
  const float* w2_1 = (const float*)d_in[6];
  const float* w1_2 = (const float*)d_in[7];
  const float* w2_2 = (const float*)d_in[8];
  const float* fc1  = (const float*)d_in[9];
  const float* fc2  = (const float*)d_in[10];
  float* out = (float*)d_out;
  _Float16* wpack = (_Float16*)d_ws;      // 276480 B of scratch

  // pack weights once per call (deterministic), then main kernel
  pack_weights_kernel<<<(PACK_ELEMS + 255) / 256, 256, 0, stream>>>(
      w1_0, w2_0, w1_1, w2_1, w1_2, w2_2, wpack);

  const int B = in_sizes[2];          // 65536 batches
  const int grid = B / BPG;           // 8 batches per workgroup -> 8192 WGs
  gcn_wmma_kernel<<<grid, THREADS, SMEM_BYTES, stream>>>(
      ops, adjm, numv, wpack, fc1, fc2, out);
}
